// RGATLayer_1219770712374
// MI455X (gfx1250) — compile-verified, matrix operand order checked by reading, not run on hardware
//
#include <hip/hip_runtime.h>
#include <hip/hip_bf16.h>
#include <stdint.h>

#define N_NODES 20000
#define IN_FEAT 512
#define HEADS 8
#define OUT_FEAT 64
#define N_REL 3
#define N_EDGES 320000
#define FEAT (HEADS * OUT_FEAT) /* 512 */
#define NEG_SLOPE 0.2f

#define LDS_STRIDE 520            /* 512 + 8 shorts: 1040 B rows, 16B aligned, bank-spread */
#define LDS_BYTES  (64 * LDS_STRIDE * 2)

typedef __attribute__((ext_vector_type(16))) __bf16 v16bf;
typedef __attribute__((ext_vector_type(8)))  __bf16 v8bf;
typedef __attribute__((ext_vector_type(8)))  float  v8f;

__device__ __forceinline__ unsigned short f32_to_bf16_rne(float f) {
    union { float f; uint32_t u; } v; v.f = f;
    uint32_t u = v.u;
    uint32_t r = u + 0x7FFFu + ((u >> 16) & 1u);
    return (unsigned short)(r >> 16);
}

// ---------------- conversion / init kernels ----------------

__global__ void k_convert_x(const float* __restrict__ x, unsigned short* __restrict__ xb, int total) {
    int i = blockIdx.x * blockDim.x + threadIdx.x;
    if (i < total) xb[i] = f32_to_bf16_rne(x[i]);
}

// W: [R, K=512, N=512] fp32  ->  Wt: [R, N=512, K=512] bf16 (K-contiguous for B-fragments)
__global__ void k_convert_Wt(const float* __restrict__ W, unsigned short* __restrict__ Wt) {
    int i = blockIdx.x * blockDim.x + threadIdx.x;
    const int total = N_REL * IN_FEAT * FEAT;
    if (i >= total) return;
    int r   = i / (IN_FEAT * FEAT);
    int rem = i - r * (IN_FEAT * FEAT);
    int n   = rem / IN_FEAT;
    int k   = rem - n * IN_FEAT;
    Wt[i] = f32_to_bf16_rne(W[(size_t)r * IN_FEAT * FEAT + (size_t)k * FEAT + n]);
}

// out[n, c] = bias0[c] + bias1[c] + bias2[c]   (HeteroGraphConv sums relations)
__global__ void k_init_out(float* __restrict__ out, const float* __restrict__ bias) {
    int i = blockIdx.x * blockDim.x + threadIdx.x;
    const int total = N_NODES * FEAT;
    if (i >= total) return;
    int c = i & (FEAT - 1);
    out[i] = bias[c] + bias[FEAT + c] + bias[2 * FEAT + c];
}

// ---------------- WMMA GEMM: Z = Xbf16 @ Wr  (fp32 accumulate) ----------------
// Block = 256 threads (8 waves) -> 128 rows x 64 cols.  The 64x512 bf16 B-panel is
// staged once per block into LDS via async global->LDS copies (ASYNCcnt), then every
// wave reads its B fragments with ds_load_b128; A fragments stream from global (L2).
__global__ void __launch_bounds__(256) k_gemm_bf16(const unsigned short* __restrict__ A,
                                                   const unsigned short* __restrict__ Bt,
                                                   float* __restrict__ Z) {
    extern __shared__ unsigned short Bl[];      // [64][LDS_STRIDE] padded panel
    const int tid  = threadIdx.x;
    const int wave = tid >> 5;
    const int lane = tid & 31;
    const int l    = lane & 15;
    const int hi   = lane >> 4;
    const int colBase = blockIdx.y * 64;

    // ---- async-stage B panel: 64 rows x 512 K bf16 = 64 KB (16 B128 copies/thread) ----
    {
        const unsigned long long gbase =
            (unsigned long long)(uintptr_t)(Bt + (size_t)colBase * IN_FEAT);
        const unsigned int lbase = (unsigned int)(uintptr_t)Bl;
        for (int u = tid; u < 64 * 64; u += 256) {
            int row = u >> 6;
            int kc  = u & 63;
            unsigned long long ga = gbase + (unsigned long long)(row * IN_FEAT + kc * 8) * 2ull;
            unsigned int       la = lbase + (unsigned int)(row * LDS_STRIDE + kc * 8) * 2u;
            asm volatile("global_load_async_to_lds_b128 %0, %1, off"
                         :: "v"(la), "v"(ga) : "memory");
        }
        asm volatile("s_wait_asynccnt 0" ::: "memory");
    }
    __syncthreads();

    const int rowTile = blockIdx.x * 8 + wave;   // wave-uniform guard: EXEC stays all-ones
    if (rowTile < N_NODES / 16) {
        const int rowBase = rowTile * 16;

        v8f acc[4];
#pragma unroll
        for (int j = 0; j < 4; ++j)
#pragma unroll
            for (int i = 0; i < 8; ++i) acc[j][i] = 0.0f;

        // A-fragment per-lane base: row (rowBase+l), K offset hi*8 (lanes 16..31 hold K+8 / K+24)
        const unsigned short* aRow = A + (size_t)(rowBase + l) * IN_FEAT + hi * 8;
        const unsigned short* bls[4];
#pragma unroll
        for (int j = 0; j < 4; ++j)
            bls[j] = Bl + (j * 16 + l) * LDS_STRIDE + hi * 8;

        for (int k0 = 0; k0 < IN_FEAT; k0 += 32) {
            v8bf alo = *(const v8bf*)(aRow + k0);        // K = k0 + hi*8 .. +7   -> elems 0..7
            v8bf ahi = *(const v8bf*)(aRow + k0 + 16);   // K = k0+16+hi*8 .. +7  -> elems 8..15
            v16bf a;
#pragma unroll
            for (int i = 0; i < 8; ++i) { a[i] = alo[i]; a[i + 8] = ahi[i]; }

#pragma unroll
            for (int j = 0; j < 4; ++j) {
                v8bf blo = *(const v8bf*)(bls[j] + k0);  // ds_load_b128
                v8bf bhi = *(const v8bf*)(bls[j] + k0 + 16);
                v16bf b;
#pragma unroll
                for (int i = 0; i < 8; ++i) { b[i] = blo[i]; b[i + 8] = bhi[i]; }
                acc[j] = __builtin_amdgcn_wmma_f32_16x16x32_bf16(
                    false, a, false, b, (short)0, acc[j], false, false);
            }
        }

        // C/D layout: VGPR v, lanes 0-15 -> M=v, N=lane; lanes 16-31 -> M=8+v, N=lane-16
#pragma unroll
        for (int j = 0; j < 4; ++j) {
            const int col = colBase + j * 16 + l;
            float* zp = Z + (size_t)(rowBase + hi * 8) * FEAT + col;
#pragma unroll
            for (int v = 0; v < 8; ++v) zp[(size_t)v * FEAT] = acc[j][v];
        }
    }
}

// ---------------- el/er + softmax-state init ----------------
__global__ void k_elr(const float* __restrict__ Z, const float* __restrict__ al,
                      const float* __restrict__ ar, float* __restrict__ el,
                      float* __restrict__ er, float* __restrict__ m, float* __restrict__ s) {
    int i = blockIdx.x * blockDim.x + threadIdx.x;
    if (i >= N_NODES * HEADS) return;
    int n = i >> 3, h = i & 7;
    const float* zp  = Z + (size_t)n * FEAT + h * OUT_FEAT;
    const float* alp = al + h * OUT_FEAT;
    const float* arp = ar + h * OUT_FEAT;
    float sl = 0.f, sr = 0.f;
#pragma unroll 8
    for (int d = 0; d < OUT_FEAT; ++d) { float z = zp[d]; sl += z * alp[d]; sr += z * arp[d]; }
    el[i] = sl; er[i] = sr;
    m[i] = -__builtin_inff();
    s[i] = 0.0f;
}

// sign-aware float atomic max via int/uint atomics (init must be -inf)
__device__ __forceinline__ void atomicMaxF(float* addr, float val) {
    if (val >= 0.0f) atomicMax((int*)addr, __float_as_int(val));
    else             atomicMin((unsigned int*)addr, __float_as_uint(val));
}

__device__ __forceinline__ float leaky(float v) { return v > 0.0f ? v : NEG_SLOPE * v; }

__global__ void k_edge_max(const int* __restrict__ src, const int* __restrict__ dst,
                           const float* __restrict__ el, const float* __restrict__ er,
                           float* __restrict__ m) {
    int i = blockIdx.x * blockDim.x + threadIdx.x;
    if (i >= N_EDGES * HEADS) return;
    int e = i >> 3, h = i & 7;
    int sN = src[e], dN = dst[e];
    float v = leaky(el[sN * 8 + h] + er[dN * 8 + h]);
    atomicMaxF(&m[dN * 8 + h], v);
}

__global__ void k_edge_exp(const int* __restrict__ src, const int* __restrict__ dst,
                           const float* __restrict__ el, const float* __restrict__ er,
                           const float* __restrict__ m, float* __restrict__ av,
                           float* __restrict__ s) {
    int i = blockIdx.x * blockDim.x + threadIdx.x;
    if (i >= N_EDGES * HEADS) return;
    int e = i >> 3, h = i & 7;
    int sN = src[e], dN = dst[e];
    float v  = leaky(el[sN * 8 + h] + er[dN * 8 + h]);
    float ex = __expf(v - m[dN * 8 + h]);
    av[i] = ex;
    atomicAdd(&s[dN * 8 + h], ex);
}

__global__ void k_norm(const int* __restrict__ dst, float* __restrict__ av,
                       const float* __restrict__ s) {
    int i = blockIdx.x * blockDim.x + threadIdx.x;
    if (i >= N_EDGES * HEADS) return;
    int e = i >> 3, h = i & 7;
    float den = s[dst[e] * 8 + h];
    av[i] = av[i] / (den > 0.0f ? den : 1.0f);
}

// one block per edge: out[dst, :] += w[h] * z[src, :]
__global__ void __launch_bounds__(256) k_agg(const int* __restrict__ src, const int* __restrict__ dst,
                                             const float* __restrict__ av, const float* __restrict__ Z,
                                             float* __restrict__ out) {
    int e  = blockIdx.x;
    int c0 = threadIdx.x * 2;                 // 256 threads x 2 floats = 512
    int sN = src[e], dN = dst[e];
    int h  = c0 >> 6;
    float w = av[(size_t)e * HEADS + h];
    const float* zp = Z + (size_t)sN * FEAT + c0;
    float* op       = out + (size_t)dN * FEAT + c0;
    float2 zv = *(const float2*)zp;
    atomicAdd(op,     w * zv.x);
    atomicAdd(op + 1, w * zv.y);
}

// ---------------- launcher ----------------
extern "C" void kernel_launch(void* const* d_in, const int* in_sizes, int n_in,
                              void* d_out, int out_size, void* d_ws, size_t ws_size,
                              hipStream_t stream) {
    (void)in_sizes; (void)n_in; (void)out_size; (void)ws_size;
    const float* x      = (const float*)d_in[0];
    const int*   edges  = (const int*)d_in[1];
    const float* W      = (const float*)d_in[2];
    const float* attn_l = (const float*)d_in[3];
    const float* attn_r = (const float*)d_in[4];
    const float* bias   = (const float*)d_in[5];
    float* out = (float*)d_out;

    char*  ws  = (char*)d_ws;
    size_t off = 0;
    auto alloc = [&](size_t bytes) -> char* {
        char* p = ws + off; off = (off + bytes + 255) & ~(size_t)255; return p;
    };
    unsigned short* xb = (unsigned short*)alloc((size_t)N_NODES * IN_FEAT * 2);
    unsigned short* Wt = (unsigned short*)alloc((size_t)N_REL * IN_FEAT * FEAT * 2);
    float* Z  = (float*)alloc((size_t)N_NODES * FEAT * 4);
    float* el = (float*)alloc((size_t)N_NODES * HEADS * 4);
    float* er = (float*)alloc((size_t)N_NODES * HEADS * 4);
    float* m  = (float*)alloc((size_t)N_NODES * HEADS * 4);
    float* s  = (float*)alloc((size_t)N_NODES * HEADS * 4);
    float* av = (float*)alloc((size_t)N_EDGES * HEADS * 4);

    {
        int total = N_NODES * IN_FEAT;
        k_convert_x<<<(total + 255) / 256, 256, 0, stream>>>(x, xb, total);
    }
    {
        int total = N_REL * IN_FEAT * FEAT;
        k_convert_Wt<<<(total + 255) / 256, 256, 0, stream>>>(W, Wt);
    }
    {
        int total = N_NODES * FEAT;
        k_init_out<<<(total + 255) / 256, 256, 0, stream>>>(out, bias);
    }

    for (int r = 0; r < N_REL; ++r) {
        const unsigned short* Wtr = Wt + (size_t)r * IN_FEAT * FEAT;
        // 128-row x 64-col tiles: grid (ceil(1250/8), 8), 8 waves/block, 65 KB dynamic LDS
        dim3 gg((N_NODES / 16 + 7) / 8, FEAT / 64);
        k_gemm_bf16<<<gg, 256, LDS_BYTES, stream>>>(xb, Wtr, Z);

        int nh = N_NODES * HEADS;
        k_elr<<<(nh + 127) / 128, 128, 0, stream>>>(
            Z, attn_l + (size_t)r * HEADS * OUT_FEAT, attn_r + (size_t)r * HEADS * OUT_FEAT,
            el, er, m, s);

        const int* srcE = edges + (size_t)r * 2 * N_EDGES;
        const int* dstE = srcE + N_EDGES;
        int eh = N_EDGES * HEADS;
        k_edge_max<<<(eh + 255) / 256, 256, 0, stream>>>(srcE, dstE, el, er, m);
        k_edge_exp<<<(eh + 255) / 256, 256, 0, stream>>>(srcE, dstE, el, er, m, av, s);
        k_norm<<<(eh + 255) / 256, 256, 0, stream>>>(dstE, av, s);
        k_agg<<<N_EDGES, 256, 0, stream>>>(srcE, dstE, av, Z, out);
    }
}
